// Fusion_Module_47261820125628
// MI455X (gfx1250) — matrix operand order repaired and checked by reading, not compiled, not run on hardware
//
#include <hip/hip_runtime.h>
#include <math.h>

// ---------------------------------------------------------------------------
// Shapes (fixed by the reference)
// ---------------------------------------------------------------------------
#define B_  4
#define C_  3
#define N_  512
#define M_  4
#define L_  64
#define O_  64
#define D_  64
#define NO_ (N_ * O_)               // 32768 = 2^15
#define PIX_ (B_ * N_ * O_)         // 131072 pixels for the 1x1-conv stages
#define YSZ_ ((size_t)B_ * D_ * N_ * O_)  // 8388608 floats
#define LDH 72                      // bf16 leading dim: 144B rows, 16B-aligned

typedef __attribute__((ext_vector_type(16))) __bf16 v16bf;
typedef __attribute__((ext_vector_type(8)))  __bf16 v8bf;
typedef __attribute__((ext_vector_type(8)))  float  v8f;

// f32 -> bf16 bits, round-to-nearest-even
__device__ __forceinline__ unsigned short bfbits(float f) {
    union { float f; unsigned u; } a; a.f = f;
    unsigned r = a.u + 0x7FFFu + ((a.u >> 16) & 1u);
    return (unsigned short)(r >> 16);
}
__device__ __forceinline__ __bf16 bfc(float f) {
    union { unsigned short s; __bf16 b; } c; c.s = bfbits(f);
    return c.b;
}
// packed store of two adjacent bf16 (p must be 4B-aligned)
__device__ __forceinline__ void store_bf2(__bf16* p, float lo, float hi) {
    const unsigned u = (unsigned)bfbits(lo) | ((unsigned)bfbits(hi) << 16);
    *(unsigned*)p = u;
}

// A-operand fragment (16x32 bf16 tile, ROW-major [m][k] in LDS, ld = LDH).
// lane<16 -> row=lane, K={0..7,16..23}; lane>=16 -> row=lane-16, K={8..15,24..31}
__device__ __forceinline__ v16bf load_a_frag(const __bf16* tile) {
    const int lane = threadIdx.x & 31;
    const __bf16* p = tile + (lane & 15) * LDH + ((lane >> 4) << 3);
    v8bf lo = *(const v8bf*)p;          // K = kb+0..7
    v8bf hi = *(const v8bf*)(p + 16);   // K = kb+16..23
    return __builtin_shufflevector(lo, hi, 0, 1, 2, 3, 4, 5, 6, 7,
                                           8, 9, 10, 11, 12, 13, 14, 15);
}

// B-operand fragment (32x16 bf16 tile) stored COL-major [n][k] in LDS, ld=LDH.
// lanes 0-15: col=lane, K={0..7,16..23} ; lanes 16-31: col=lane-16, K={8..15,24..31}
__device__ __forceinline__ v16bf load_b_frag(const __bf16* tile) {
    const int lane = threadIdx.x & 31;
    const __bf16* p = tile + (lane & 15) * LDH + ((lane >> 4) << 3);
    v8bf lo = *(const v8bf*)p;
    v8bf hi = *(const v8bf*)(p + 16);
    return __builtin_shufflevector(lo, hi, 0, 1, 2, 3, 4, 5, 6, 7,
                                           8, 9, 10, 11, 12, 13, 14, 15);
}

__device__ __forceinline__ v8f wmma_bf16(v16bf a, v16bf b, v8f c) {
    return __builtin_amdgcn_wmma_f32_16x16x32_bf16(false, a, false, b,
                                                   (short)0, c, false, false);
}

// ---------------------------------------------------------------------------
// Kernel 1: fused QKV projection + attention + softmax + V*A + M-combine.
// One block per (b, n); 128 threads = 4 waves; wave w owns output rows
// [16w, 16w+16). A (softmax) goes straight to d_out; Ocomb accumulates over m
// in registers (combine weight folded into the bf16 A operand) -> 1 HBM write.
// ---------------------------------------------------------------------------
__global__ __launch_bounds__(128) void attn_kernel(
    const float* __restrict__ x, const float* __restrict__ x_date,
    const float* __restrict__ y_date, const float* __restrict__ top_value,
    const float* __restrict__ Wq, const float* __restrict__ bq,
    const float* __restrict__ Wk, const float* __restrict__ bk,
    const float* __restrict__ Wv, const float* __restrict__ bv,
    const float* __restrict__ wsum,
    float* __restrict__ A_out, float* __restrict__ Ocomb)
{
    const int bn   = blockIdx.x;
    const int b    = bn / N_;
    const int n    = bn % N_;
    const int tid  = threadIdx.x;
    const int wave = tid >> 5;
    const int lane = tid & 31;

    __shared__ float  sWq[192], sWk[192], sWv[192];
    __shared__ float  sbq[64], sbk[64], sbv[64];
    __shared__ float  xs[3][64], ys[3][64], vs[3][64];
    __shared__ __bf16 qT[64 * LDH];   // [l][d]  row-major  (A of scores)
    __shared__ __bf16 kT[64 * LDH];   // [o][d]  col-major  (B of scores)
    __shared__ __bf16 vM[64 * LDH];   // [d][l]  row-major  (A of V*A)
    __shared__ __bf16 aT[64 * LDH];   // [o][l]  col-major  (B of V*A)
    __shared__ float  sc[64 * 65];    // scores / exp scratch

    if (tid < 192) { sWq[tid] = Wq[tid]; sWk[tid] = Wk[tid]; sWv[tid] = Wv[tid]; }
    if (tid < 64)  { sbq[tid] = bq[tid]; sbk[tid] = bk[tid]; sbv[tid] = bv[tid]; }

    v8f acc[4];
#pragma unroll
    for (int t = 0; t < 4; ++t)
#pragma unroll
        for (int i = 0; i < 8; ++i) acc[t][i] = 0.0f;

    const int rb    = wave * 16;
    const int col16 = lane & 15;
    const int r8    = (lane >> 4) << 3;

    for (int m = 0; m < M_; ++m) {
        __syncthreads();
        // ---- stage raw C=3 slices + prefetch next m -------------------
        if (tid < 192) {
            const int c = tid >> 6, e = tid & 63;
            const size_t base = ((((size_t)b * C_ + c) * N_ + n) * M_ + m) * 64 + e;
            xs[c][e] = x_date[base];
            ys[c][e] = y_date[base];
            vs[c][e] = x[base];
            if (m + 1 < M_) {                       // global_prefetch_b8
                __builtin_prefetch(&x_date[base + 64]);
                __builtin_prefetch(&y_date[base + 64]);
                __builtin_prefetch(&x[base + 64]);
            }
        }
        __syncthreads();
        // ---- 1x1x1 conv projections (C=3 -> D=64), packed bf16 stores -
        // qT[l][d]: contiguous in d -> pair over d
        for (int i = tid; i < 2048; i += 128) {
            const int l = i >> 5, dp = (i & 31) << 1;
            const float x0v = xs[0][l], x1v = xs[1][l], x2v = xs[2][l];
            const float qa = sbq[dp]     + sWq[dp * 3 + 0] * x0v
                                         + sWq[dp * 3 + 1] * x1v
                                         + sWq[dp * 3 + 2] * x2v;
            const float qb = sbq[dp + 1] + sWq[dp * 3 + 3] * x0v
                                         + sWq[dp * 3 + 4] * x1v
                                         + sWq[dp * 3 + 5] * x2v;
            store_bf2(qT + l * LDH + dp, qa, qb);
        }
        // kT[o][d]: contiguous in d -> pair over d
        for (int i = tid; i < 2048; i += 128) {
            const int o = i >> 5, dp = (i & 31) << 1;
            const float y0v = ys[0][o], y1v = ys[1][o], y2v = ys[2][o];
            const float ka = sbk[dp]     + sWk[dp * 3 + 0] * y0v
                                         + sWk[dp * 3 + 1] * y1v
                                         + sWk[dp * 3 + 2] * y2v;
            const float kb = sbk[dp + 1] + sWk[dp * 3 + 3] * y0v
                                         + sWk[dp * 3 + 4] * y1v
                                         + sWk[dp * 3 + 5] * y2v;
            store_bf2(kT + o * LDH + dp, ka, kb);
        }
        // vM[d][l]: contiguous in l -> pair over l
        for (int i = tid; i < 2048; i += 128) {
            const int d = i >> 5, lp = (i & 31) << 1;
            const float w0 = sWv[d * 3 + 0], w1 = sWv[d * 3 + 1],
                        w2 = sWv[d * 3 + 2], bb = sbv[d];
            const float va = bb + w0 * vs[0][lp]     + w1 * vs[1][lp]
                                + w2 * vs[2][lp];
            const float vb = bb + w0 * vs[0][lp + 1] + w1 * vs[1][lp + 1]
                                + w2 * vs[2][lp + 1];
            store_bf2(vM + d * LDH + lp, va, vb);
        }
        __syncthreads();
        // ---- scores = Q^T K  (batch all B-fragments, then WMMA chain) -
        {
            const v16bf qa0 = load_a_frag(qT + rb * LDH);
            const v16bf qa1 = load_a_frag(qT + rb * LDH + 32);
            v16bf bfr[8];
#pragma unroll
            for (int ct = 0; ct < 4; ++ct) {
                bfr[2 * ct]     = load_b_frag(kT + ct * 16 * LDH);
                bfr[2 * ct + 1] = load_b_frag(kT + ct * 16 * LDH + 32);
            }
#pragma unroll
            for (int ct = 0; ct < 4; ++ct) {
                v8f cc;
#pragma unroll
                for (int i = 0; i < 8; ++i) cc[i] = 0.0f;
                cc = wmma_bf16(qa0, bfr[2 * ct], cc);
                cc = wmma_bf16(qa1, bfr[2 * ct + 1], cc);
                const int co = ct * 16 + col16;
#pragma unroll
                for (int r = 0; r < 8; ++r)
                    sc[(rb + r8 + r) * 65 + co] = cc[r] * 0.125f;  // 1/sqrt(D)
            }
        }
        __syncthreads();
        // ---- softmax over L (per column o), emit A + weighted bf16 A --
        const float wnm = wsum[m] * (m == 0 ? 1.0f : top_value[n * (M_ - 1) + (m - 1)]);
        if (tid < 64) {
            const int o = tid;
            float mx = -3.4e38f;
            for (int l = 0; l < 64; ++l) mx = fmaxf(mx, sc[l * 65 + o]);
            float sum = 0.0f;
            for (int l = 0; l < 64; ++l) {
                float e = expf(sc[l * 65 + o] - mx);
                sc[l * 65 + o] = e;
                sum += e;
            }
            const float inv = 1.0f / sum;
            const size_t abase = ((((size_t)b * N_ + n) * M_ + m) * (size_t)L_) * O_ + o;
            for (int l = 0; l < 64; l += 2) {
                const float a0 = sc[l * 65 + o] * inv;
                const float a1 = sc[(l + 1) * 65 + o] * inv;
                A_out[abase + (size_t)l * O_]       = a0;   // unscaled A output
                A_out[abase + (size_t)(l + 1) * O_] = a1;
                store_bf2(aT + o * LDH + l, a0 * wnm, a1 * wnm);
            }
        }
        __syncthreads();
        // ---- Ocomb += V * (A * w)  (contraction over L=64) ------------
        {
            const v16bf va0 = load_a_frag(vM + rb * LDH);
            const v16bf va1 = load_a_frag(vM + rb * LDH + 32);
            v16bf bfr[8];
#pragma unroll
            for (int ct = 0; ct < 4; ++ct) {
                bfr[2 * ct]     = load_b_frag(aT + ct * 16 * LDH);
                bfr[2 * ct + 1] = load_b_frag(aT + ct * 16 * LDH + 32);
            }
#pragma unroll
            for (int ct = 0; ct < 4; ++ct) {
                acc[ct] = wmma_bf16(va0, bfr[2 * ct], acc[ct]);
                acc[ct] = wmma_bf16(va1, bfr[2 * ct + 1], acc[ct]);
            }
        }
    }
    // ---- write Ocomb[b][d][n][o] once ---------------------------------
#pragma unroll
    for (int ct = 0; ct < 4; ++ct) {
        const int o = ct * 16 + col16;
#pragma unroll
        for (int r = 0; r < 8; ++r) {
            const int d = rb + r8 + r;
            Ocomb[(((size_t)b * D_ + d) * N_ + n) * O_ + o] = acc[ct][r];
        }
    }
}

// ---------------------------------------------------------------------------
// Kernel 2: per-channel mean / rstd over (B, N, O). One block per channel.
// Deterministic tree reduction (no float atomics -> bit-stable replays).
// ---------------------------------------------------------------------------
__global__ __launch_bounds__(256) void stats_kernel(
    const float* __restrict__ src, float* __restrict__ mean, float* __restrict__ rstd)
{
    const int d = blockIdx.x;
    const int tid = threadIdx.x;
    float s = 0.0f, q = 0.0f;
    for (int i = tid; i < PIX_; i += 256) {
        const int b  = i >> 15;
        const int qq = i & (NO_ - 1);
        const float v = src[(((size_t)(b * D_ + d)) << 15) + qq];
        s += v; q += v * v;
    }
    __shared__ float rs[256], rq[256];
    rs[tid] = s; rq[tid] = q;
    __syncthreads();
    for (int off = 128; off > 0; off >>= 1) {
        if (tid < off) { rs[tid] += rs[tid + off]; rq[tid] += rq[tid + off]; }
        __syncthreads();
    }
    if (tid == 0) {
        const float mu  = rs[0] * (1.0f / PIX_);
        const float var = rq[0] * (1.0f / PIX_) - mu * mu;   // biased var
        mean[d] = mu;
        rstd[d] = rsqrtf(var + 1e-5f);
    }
}

// ---------------------------------------------------------------------------
// Kernel 3: t = x0 + W2 * gelu(W1 * x0 + b1) + b2, x0 = BN1(Ocomb).
// One block = 64 pixels; D x D matmuls via bf16 WMMA.
// ---------------------------------------------------------------------------
__global__ __launch_bounds__(128) void mlp1_kernel(
    const float* __restrict__ Ocomb,
    const float* __restrict__ W1, const float* __restrict__ b1,
    const float* __restrict__ W2, const float* __restrict__ b2,
    const float* __restrict__ g1, const float* __restrict__ be1,
    const float* __restrict__ mean1, const float* __restrict__ rstd1,
    float* __restrict__ tbuf)
{
    const int tid  = threadIdx.x;
    const int wave = tid >> 5;
    const int lane = tid & 31;
    const int pix0 = blockIdx.x * 64;
    const int b    = pix0 >> 15;
    const int q0   = pix0 & (NO_ - 1);

    __shared__ __bf16 W1b[64 * LDH], W2b[64 * LDH];  // [row][k] row-major
    __shared__ __bf16 x0t[64 * LDH], ht[64 * LDH];   // [pixel][chan] col-major
    __shared__ float  x0f[64 * 65];
    __shared__ float  sb1[64], sb2[64], sg[64], sbe[64], sm[64], sr[64];

    // weights: contiguous in k -> packed pair stores
    for (int i = tid; i < 2048; i += 128) {
        const int r = i >> 5, cp = (i & 31) << 1;
        store_bf2(W1b + r * LDH + cp, W1[r * 64 + cp], W1[r * 64 + cp + 1]);
        store_bf2(W2b + r * LDH + cp, W2[r * 64 + cp], W2[r * 64 + cp + 1]);
    }
    if (tid < 64) {
        sb1[tid] = b1[tid]; sb2[tid] = b2[tid];
        sg[tid] = g1[tid];  sbe[tid] = be1[tid];
        sm[tid] = mean1[tid]; sr[tid] = rstd1[tid];
    }
    __syncthreads();
    // x0 tile (coalesced global reads along pixel dim j)
    for (int i = tid; i < 4096; i += 128) {
        const int c = i >> 6, j = i & 63;
        const float v  = Ocomb[(((size_t)(b * D_ + c)) << 15) + q0 + j];
        const float xn = (v - sm[c]) * sr[c] * sg[c] + sbe[c];
        x0f[c * 65 + j]  = xn;        // residual (row = channel)
        x0t[j * LDH + c] = bfc(xn);   // col-major B operand
    }
    __syncthreads();

    const int rb    = wave * 16;
    const int col16 = lane & 15;
    const int r8    = (lane >> 4) << 3;

    // h = gelu(W1 * x0 + b1)
    {
        const v16bf wa0 = load_a_frag(W1b + rb * LDH);
        const v16bf wa1 = load_a_frag(W1b + rb * LDH + 32);
        v16bf bfr[8];
#pragma unroll
        for (int ct = 0; ct < 4; ++ct) {
            bfr[2 * ct]     = load_b_frag(x0t + ct * 16 * LDH);
            bfr[2 * ct + 1] = load_b_frag(x0t + ct * 16 * LDH + 32);
        }
#pragma unroll
        for (int ct = 0; ct < 4; ++ct) {
            v8f cc;
#pragma unroll
            for (int i = 0; i < 8; ++i) cc[i] = 0.0f;
            cc = wmma_bf16(wa0, bfr[2 * ct], cc);
            cc = wmma_bf16(wa1, bfr[2 * ct + 1], cc);
            const int co = ct * 16 + col16;
#pragma unroll
            for (int r = 0; r < 8; r += 2) {     // rows contiguous -> pair
                const int row = rb + r8 + r;
                float h0 = cc[r] + sb1[row];
                float h1 = cc[r + 1] + sb1[row + 1];
                h0 = 0.5f * h0 * (1.0f + erff(h0 * 0.70710678118654752f));
                h1 = 0.5f * h1 * (1.0f + erff(h1 * 0.70710678118654752f));
                store_bf2(ht + co * LDH + row, h0, h1);
            }
        }
    }
    __syncthreads();
    // t = x0 + W2 * h + b2
    {
        const v16bf wa0 = load_a_frag(W2b + rb * LDH);
        const v16bf wa1 = load_a_frag(W2b + rb * LDH + 32);
        v16bf bfr[8];
#pragma unroll
        for (int ct = 0; ct < 4; ++ct) {
            bfr[2 * ct]     = load_b_frag(ht + ct * 16 * LDH);
            bfr[2 * ct + 1] = load_b_frag(ht + ct * 16 * LDH + 32);
        }
#pragma unroll
        for (int ct = 0; ct < 4; ++ct) {
            v8f cc;
#pragma unroll
            for (int i = 0; i < 8; ++i) cc[i] = 0.0f;
            cc = wmma_bf16(wa0, bfr[2 * ct], cc);
            cc = wmma_bf16(wa1, bfr[2 * ct + 1], cc);
            const int co = ct * 16 + col16;
#pragma unroll
            for (int r = 0; r < 8; ++r) {
                const int row = rb + r8 + r;
                tbuf[(((size_t)(b * D_ + row)) << 15) + q0 + co] =
                    cc[r] + sb2[row] + x0f[row * 65 + co];
            }
        }
    }
}

// ---------------------------------------------------------------------------
// Kernel 4: z = Wf * BN2(t) + bf
// ---------------------------------------------------------------------------
__global__ __launch_bounds__(128) void mlp2_kernel(
    const float* __restrict__ tbuf,
    const float* __restrict__ Wf, const float* __restrict__ bf_bias,
    const float* __restrict__ g2, const float* __restrict__ be2,
    const float* __restrict__ mean2, const float* __restrict__ rstd2,
    float* __restrict__ zbuf)
{
    const int tid  = threadIdx.x;
    const int wave = tid >> 5;
    const int lane = tid & 31;
    const int pix0 = blockIdx.x * 64;
    const int b    = pix0 >> 15;
    const int q0   = pix0 & (NO_ - 1);

    __shared__ __bf16 Wfb[64 * LDH];   // [row][k]
    __shared__ __bf16 ut[64 * LDH];    // [pixel][chan]
    __shared__ float  sbf[64], sg[64], sbe[64], sm[64], sr[64];

    for (int i = tid; i < 2048; i += 128) {
        const int r = i >> 5, cp = (i & 31) << 1;
        store_bf2(Wfb + r * LDH + cp, Wf[r * 64 + cp], Wf[r * 64 + cp + 1]);
    }
    if (tid < 64) {
        sbf[tid] = bf_bias[tid];
        sg[tid] = g2[tid]; sbe[tid] = be2[tid];
        sm[tid] = mean2[tid]; sr[tid] = rstd2[tid];
    }
    __syncthreads();
    for (int i = tid; i < 4096; i += 128) {
        const int c = i >> 6, j = i & 63;
        const float v = tbuf[(((size_t)(b * D_ + c)) << 15) + q0 + j];
        ut[j * LDH + c] = bfc((v - sm[c]) * sr[c] * sg[c] + sbe[c]);
    }
    __syncthreads();

    const int rb    = wave * 16;
    const int col16 = lane & 15;
    const int r8    = (lane >> 4) << 3;

    const v16bf wa0 = load_a_frag(Wfb + rb * LDH);
    const v16bf wa1 = load_a_frag(Wfb + rb * LDH + 32);
    v16bf bfr[8];
#pragma unroll
    for (int ct = 0; ct < 4; ++ct) {
        bfr[2 * ct]     = load_b_frag(ut + ct * 16 * LDH);
        bfr[2 * ct + 1] = load_b_frag(ut + ct * 16 * LDH + 32);
    }
#pragma unroll
    for (int ct = 0; ct < 4; ++ct) {
        v8f cc;
#pragma unroll
        for (int i = 0; i < 8; ++i) cc[i] = 0.0f;
        cc = wmma_bf16(wa0, bfr[2 * ct], cc);
        cc = wmma_bf16(wa1, bfr[2 * ct + 1], cc);
        const int co = ct * 16 + col16;
#pragma unroll
        for (int r = 0; r < 8; ++r) {
            const int row = rb + r8 + r;
            zbuf[(((size_t)(b * D_ + row)) << 15) + q0 + co] = cc[r] + sbf[row];
        }
    }
}

// ---------------------------------------------------------------------------
// Kernel 5: y = BN3(z)  (elementwise, writes final y region of d_out)
// ---------------------------------------------------------------------------
__global__ __launch_bounds__(256) void bn_out_kernel(
    const float* __restrict__ z,
    const float* __restrict__ g3, const float* __restrict__ be3,
    const float* __restrict__ mean3, const float* __restrict__ rstd3,
    float* __restrict__ y)
{
    const size_t idx = (size_t)blockIdx.x * 256 + threadIdx.x;
    if (idx < YSZ_) {
        const int d = (int)((idx >> 15) & 63);
        y[idx] = (z[idx] - mean3[d]) * rstd3[d] * g3[d] + be3[d];
    }
}

// ---------------------------------------------------------------------------
// Host launcher. Input order = setup_inputs() dict order, params flattened
// in insertion order: x, x_date, y_date, top_value, Wq,bq,Wk,bk,Wv,bv,
// weight_sum, W1,b1,W2,b2,Wf,bf, g1,be1,g2,be2,g3,be3.
// d_out = [ y (8388608 f32) | A (33554432 f32) ].
// ws: Ocomb (33.5MB) | tbuf (33.5MB) | stats (6*64 f32); zbuf reuses Ocomb.
// ---------------------------------------------------------------------------
extern "C" void kernel_launch(void* const* d_in, const int* in_sizes, int n_in,
                              void* d_out, int out_size, void* d_ws, size_t ws_size,
                              hipStream_t stream)
{
    (void)in_sizes; (void)n_in; (void)out_size; (void)ws_size;
    const float* x    = (const float*)d_in[0];
    const float* xd   = (const float*)d_in[1];
    const float* yd   = (const float*)d_in[2];
    const float* tv   = (const float*)d_in[3];
    const float* Wq   = (const float*)d_in[4];
    const float* bq   = (const float*)d_in[5];
    const float* Wk   = (const float*)d_in[6];
    const float* bk   = (const float*)d_in[7];
    const float* Wv   = (const float*)d_in[8];
    const float* bv   = (const float*)d_in[9];
    const float* wsum = (const float*)d_in[10];
    const float* W1   = (const float*)d_in[11];
    const float* b1   = (const float*)d_in[12];
    const float* W2   = (const float*)d_in[13];
    const float* b2   = (const float*)d_in[14];
    const float* Wf   = (const float*)d_in[15];
    const float* bfp  = (const float*)d_in[16];
    const float* g1   = (const float*)d_in[17];
    const float* be1  = (const float*)d_in[18];
    const float* g2   = (const float*)d_in[19];
    const float* be2  = (const float*)d_in[20];
    const float* g3   = (const float*)d_in[21];
    const float* be3  = (const float*)d_in[22];

    float* y_out = (float*)d_out;
    float* A_out = y_out + YSZ_;

    float* Ocomb = (float*)d_ws;
    float* tbuf  = Ocomb + YSZ_;
    float* stats = tbuf + YSZ_;
    float* mean1 = stats + 0,   *rstd1 = stats + 64;
    float* mean2 = stats + 128, *rstd2 = stats + 192;
    float* mean3 = stats + 256, *rstd3 = stats + 320;
    float* zbuf  = Ocomb;   // Ocomb dead after mlp1

    attn_kernel<<<dim3(B_ * N_), dim3(128), 0, stream>>>(
        x, xd, yd, tv, Wq, bq, Wk, bk, Wv, bv, wsum, A_out, Ocomb);

    stats_kernel<<<dim3(D_), dim3(256), 0, stream>>>(Ocomb, mean1, rstd1);

    mlp1_kernel<<<dim3(PIX_ / 64), dim3(128), 0, stream>>>(
        Ocomb, W1, b1, W2, b2, g1, be1, mean1, rstd1, tbuf);

    stats_kernel<<<dim3(D_), dim3(256), 0, stream>>>(tbuf, mean2, rstd2);

    mlp2_kernel<<<dim3(PIX_ / 64), dim3(128), 0, stream>>>(
        tbuf, Wf, bfp, g2, be2, mean2, rstd2, zbuf);

    stats_kernel<<<dim3(D_), dim3(256), 0, stream>>>(zbuf, mean3, rstd3);

    bn_out_kernel<<<dim3((unsigned)((YSZ_ + 255) / 256)), dim3(256), 0, stream>>>(
        zbuf, g3, be3, mean3, rstd3, y_out);
}
